// SequentialGraniteMoeHybridMoE_46780783788487
// MI455X (gfx1250) — matrix operand order repaired and checked by reading, not compiled, not run on hardware
//
#include <hip/hip_runtime.h>
#include <hip/hip_bf16.h>
#include <stdint.h>

// ---------------------------------------------------------------------------
// Fused MoE (all-experts path) for MI455X / gfx1250.
//   T=4096 tokens, H=1024, E=8, I=2048, top_k=2.
// bf16 WMMA compute (16x16x32, f32 accum), weights pre-packed to WMMA K-panel
// layout, staged into double-buffered LDS with GLOBAL_LOAD_ASYNC_TO_LDS.
// ---------------------------------------------------------------------------

typedef __attribute__((ext_vector_type(4)))  float  f32x4;
typedef __attribute__((ext_vector_type(8)))  float  v8f;
typedef __attribute__((ext_vector_type(8)))  __bf16 v8bf;
typedef __attribute__((ext_vector_type(16))) __bf16 v16bf;
typedef __attribute__((ext_vector_type(4)))  __bf16 bf4;

#define T_TOK 4096
#define Hd    1024
#define Id    2048
#define Ed    8
#define MT    32          // tokens per workgroup (2 row-tiles of 16)

// ---- CDNA5 async global->LDS DMA (ASYNCcnt) --------------------------------
__device__ __forceinline__ unsigned lds_lo32(const void* p) {
  return (unsigned)(uintptr_t)p;     // LDS aperture: low 32 bits = LDS offset
}
__device__ __forceinline__ void async_b128(unsigned lds, const void* g) {
  asm volatile("global_load_async_to_lds_b128 %0, %1, off"
               :: "v"(lds), "v"((unsigned long long)(uintptr_t)g) : "memory");
}
__device__ __forceinline__ void async_b64(unsigned lds, const void* g) {
  asm volatile("global_load_async_to_lds_b64 %0, %1, off"
               :: "v"(lds), "v"((unsigned long long)(uintptr_t)g) : "memory");
}

__device__ __forceinline__ v16bf cat8(v8bf lo, v8bf hi) {
  return __builtin_shufflevector(lo, hi, 0,1,2,3,4,5,6,7,8,9,10,11,12,13,14,15);
}

// ---------------------------------------------------------------------------
// Kernel 1: x (f32) -> xb (bf16)
// ---------------------------------------------------------------------------
__global__ void x_to_bf16_kernel(const float* __restrict__ x,
                                 __bf16* __restrict__ xb, int n4) {
  int i = blockIdx.x * blockDim.x + threadIdx.x;
  if (i < n4) {
    f32x4 v = ((const f32x4*)x)[i];
    ((bf4*)xb)[i] = __builtin_convertvector(v, bf4);
  }
}

// ---------------------------------------------------------------------------
// Kernel 2a/2b: one-time pack of weights: f32 -> bf16 WMMA K-panel layout
//   W_in  [E][H=1024][2I=4096] -> [e][kt=H/32][n=4096][kk=32]
//   W_out [E][I=2048][H=1024]  -> [e][kt=I/32][n=1024][kk=32]
// After this, every staged tile is one contiguous DMA, already transposed.
// ---------------------------------------------------------------------------
__global__ void pack_win_kernel(const float* __restrict__ w,
                                __bf16* __restrict__ p) {
  size_t i = (size_t)blockIdx.x * 256 + threadIdx.x;   // source order (coalesced)
  int n = (int)(i & 4095);
  int k = (int)((i >> 12) & 1023);
  int e = (int)(i >> 22);
  size_t d = ((((size_t)e * 32 + (k >> 5)) * 4096 + n) << 5) + (k & 31);
  p[d] = (__bf16)w[i];
}

__global__ void pack_wout_kernel(const float* __restrict__ w,
                                 __bf16* __restrict__ p) {
  size_t i = (size_t)blockIdx.x * 256 + threadIdx.x;
  int n = (int)(i & 1023);
  int k = (int)((i >> 10) & 2047);
  int e = (int)(i >> 21);
  size_t d = ((((size_t)e * 64 + (k >> 5)) * 1024 + n) << 5) + (k & 31);
  p[d] = (__bf16)w[i];
}

// ---------------------------------------------------------------------------
// Kernel 3: router -> dense top-2 softmax gates [T, E]
// ---------------------------------------------------------------------------
__global__ void moe_router_kernel(const float* __restrict__ x,
                                  const float* __restrict__ Wg,
                                  float* __restrict__ gates) {
  __shared__ float logit_lds[32][Ed];
  const int tid = threadIdx.x;
  const int tl  = tid >> 3;
  const int e   = tid & 7;
  const int t   = blockIdx.x * 32 + tl;

  const float* xr = x + (size_t)t * Hd;
  float acc = 0.0f;
  for (int h = 0; h < Hd; ++h) acc += xr[h] * Wg[h * Ed + e];
  logit_lds[tl][e] = acc;
  __syncthreads();

  if (e == 0) {
    float l1 = -1e30f, l2 = -1e30f; int i1 = -1, i2 = -1;
    for (int k = 0; k < Ed; ++k) {            // strict '>' => first index on tie
      float v = logit_lds[tl][k];
      if (v > l1)      { l2 = l1; i2 = i1; l1 = v; i1 = k; }
      else if (v > l2) { l2 = v;  i2 = k; }
    }
    float g1 = 1.0f / (1.0f + __expf(l2 - l1));
    float g2 = 1.0f - g1;
    float* gr = gates + (size_t)t * Ed;
    for (int k = 0; k < Ed; ++k)
      gr[k] = (k == i1) ? g1 : ((k == i2) ? g2 : 0.0f);
  }
}

// ---------------------------------------------------------------------------
// Kernel 4: fused all-expert gated FFN + gate-weighted combine.
//   32 tokens/WG, 8 wave32s as 2 row-groups x 4 col-groups.
//   Per expert, per I-quarter:
//     phase 1: a_lds[32][512] = gate * silu(x@Wg_in) * (x@Wu_in)  (bf16)
//     phase 2: accY[16] (128 VGPRs/lane) += a_lds @ W_out-quarter
//   All weight/x tiles arrive via async DMA into double-buffered LDS.
// ---------------------------------------------------------------------------
__launch_bounds__(256)
__global__ void moe_ffn_fused_kernel(const __bf16* __restrict__ xb,
                                     const __bf16* __restrict__ WinP,
                                     const __bf16* __restrict__ WoutP,
                                     const float* __restrict__ gates,
                                     float* __restrict__ out) {
  __shared__ __attribute__((aligned(16))) __bf16 a_lds[MT * 512];   // 32 KB
  __shared__ __attribute__((aligned(16))) __bf16 wt[2][128 * 32];   // 2 x 8 KB
  __shared__ __attribute__((aligned(16))) __bf16 xs[2][32 * 32];    // 2 x 2 KB
  __shared__ float gate_lds[MT * Ed];                               // 1 KB

  const int tid  = threadIdx.x;
  const int lane = tid & 31;
  const int wid  = tid >> 5;
  const int rw   = wid >> 2;         // row-group 0/1  (16 tokens each)
  const int wc   = wid & 3;          // col-group 0..3 (16 cols each)
  const int t0   = blockIdx.x * MT;

  gate_lds[tid] = gates[(size_t)t0 * Ed + tid];     // 256 == MT*Ed
  __syncthreads();

  const int c    = lane & 15;
  const int hi16 = lane >> 4;
  const int kbB  = hi16 << 4;        // B frag K-base 0/16
  const int ksA  = hi16 << 3;        // unused; A frag uses kbB too

  const unsigned wt0  = lds_lo32(&wt[0][0]);
  const unsigned xs0  = lds_lo32(&xs[0][0]);
  const unsigned ldsW = (unsigned)tid * 16;   // this thread's slice of a tile
  const unsigned ldsX = (unsigned)tid * 8;
  const int xrow = tid >> 3;                  // x-slab row this thread copies
  const int xcol = (tid & 7) * 4;             // 4 bf16 = 8 bytes

  v8f accY[16];                               // persistent gated output
  #pragma unroll
  for (int j = 0; j < 16; ++j)
    #pragma unroll
    for (int r = 0; r < 8; ++r) accY[j][r] = 0.0f;

  for (int e = 0; e < Ed; ++e) {
    for (int q = 0; q < 4; ++q) {             // I-quarter: cols [q*512, +512)
      // ================= Phase 1: build a_lds for this quarter ==============
      for (int ic = 0; ic < 8; ++ic) {        // 64 g-cols (+64 u-cols) / chunk
        const int g0 = q * 512 + ic * 64;
        v8f accG, accU;
        #pragma unroll
        for (int r = 0; r < 8; ++r) { accG[r] = 0.0f; accU[r] = 0.0f; }

        // prologue: DMA k-slab 0 into buffer 0
        {
          const size_t pan = (((size_t)e * 32 + 0) * 4096 + g0) << 5;
          async_b128(wt0 + ldsW,        WinP + pan + (size_t)tid * 8);
          async_b128(wt0 + 4096 + ldsW, WinP + pan + ((size_t)2048 << 5) + (size_t)tid * 8);
          async_b64 (xs0 + ldsX,        xb + (size_t)(t0 + xrow) * Hd + xcol);
        }
        int p = 0;
        for (int k0 = 0; k0 < Hd; k0 += 32) {
          if (k0 + 32 < Hd) {                 // DMA next slab into other buffer
            const int kt = (k0 + 32) >> 5;
            const size_t pan = (((size_t)e * 32 + kt) * 4096 + g0) << 5;
            async_b128(wt0 + (unsigned)(p ^ 1) * 8192 + ldsW,
                       WinP + pan + (size_t)tid * 8);
            async_b128(wt0 + (unsigned)(p ^ 1) * 8192 + 4096 + ldsW,
                       WinP + pan + ((size_t)2048 << 5) + (size_t)tid * 8);
            async_b64 (xs0 + (unsigned)(p ^ 1) * 2048 + ldsX,
                       xb + (size_t)(t0 + xrow) * Hd + (k0 + 32) + xcol);
            asm volatile("s_wait_asynccnt 0x3" ::: "memory");  // buf p complete
          } else {
            asm volatile("s_wait_asynccnt 0x0" ::: "memory");
          }
          __syncthreads();

          const __bf16* xr = &xs[p][(rw * 16 + c) * 32 + kbB];
          v16bf A = cat8(*(const v8bf*)xr, *(const v8bf*)(xr + 8));
          const __bf16* bg = &wt[p][(wc * 16 + c) * 32 + kbB];
          v16bf Bg = cat8(*(const v8bf*)bg, *(const v8bf*)(bg + 8));
          const __bf16* bu = bg + 64 * 32;
          v16bf Bu = cat8(*(const v8bf*)bu, *(const v8bf*)(bu + 8));

          accG = __builtin_amdgcn_wmma_f32_16x16x32_bf16(
                     false, A, false, Bg, (short)0, accG, false, false);
          accU = __builtin_amdgcn_wmma_f32_16x16x32_bf16(
                     false, A, false, Bu, (short)0, accU, false, false);
          __syncthreads();                    // buf p free for re-DMA
          p ^= 1;
        }

        // a = gate * silu(g) * u  (gate folded in => accY sums directly)
        #pragma unroll
        for (int r = 0; r < 8; ++r) {
          int   row = rw * 16 + r + (hi16 << 3);
          float gv  = accG[r];
          float s   = gv / (1.0f + __expf(-gv));
          float val = s * accU[r] * gate_lds[row * Ed + e];
          a_lds[row * 512 + ic * 64 + wc * 16 + c] = (__bf16)val;
        }
      } // ic
      __syncthreads();                        // a_lds complete

      // ================= Phase 2: accY += a_lds @ W_out-quarter =============
      #pragma unroll
      for (int j = 0; j < 16; ++j) {          // output col-chunk of 64
        {
          const int kt = (q * 512) >> 5;
          async_b128(wt0 + ldsW,
                     WoutP + ((((size_t)e * 64 + kt) * 1024 + j * 64) << 5)
                           + (size_t)tid * 8);
        }
        int p = 0;
        for (int k0 = 0; k0 < 512; k0 += 32) {
          if (k0 + 32 < 512) {
            const int kt = (q * 512 + k0 + 32) >> 5;
            async_b128(wt0 + (unsigned)(p ^ 1) * 8192 + ldsW,
                       WoutP + ((((size_t)e * 64 + kt) * 1024 + j * 64) << 5)
                             + (size_t)tid * 8);
            asm volatile("s_wait_asynccnt 0x1" ::: "memory");
          } else {
            asm volatile("s_wait_asynccnt 0x0" ::: "memory");
          }
          __syncthreads();

          const __bf16* ar = &a_lds[(rw * 16 + c) * 512 + k0 + kbB];
          v16bf A2 = cat8(*(const v8bf*)ar, *(const v8bf*)(ar + 8));
          const __bf16* bp = &wt[p][(wc * 16 + c) * 32 + kbB];
          v16bf B2 = cat8(*(const v8bf*)bp, *(const v8bf*)(bp + 8));

          accY[j] = __builtin_amdgcn_wmma_f32_16x16x32_bf16(
                        false, A2, false, B2, (short)0, accY[j], false, false);
          __syncthreads();
          p ^= 1;
        }
      } // j
    } // q
  } // e

  // ---- write gated, expert-summed output (WG owns its 32 rows) -------------
  #pragma unroll
  for (int j = 0; j < 16; ++j)
    #pragma unroll
    for (int r = 0; r < 8; ++r) {
      int row = rw * 16 + r + (hi16 << 3);
      out[(size_t)(t0 + row) * Hd + j * 64 + wc * 16 + c] = accY[j][r];
    }
}

// ---------------------------------------------------------------------------
extern "C" void kernel_launch(void* const* d_in, const int* in_sizes, int n_in,
                              void* d_out, int out_size, void* d_ws, size_t ws_size,
                              hipStream_t stream) {
  const float* x     = (const float*)d_in[0];   // [T,1024]
  const float* Wg    = (const float*)d_in[1];   // [1024,8]
  const float* W_in  = (const float*)d_in[2];   // [8,1024,4096]
  const float* W_out = (const float*)d_in[3];   // [8,2048,1024]
  // d_in[4] = top_k (==2, baked in)
  float* out = (float*)d_out;

  // workspace: xb 8MB | WinP 64MB | WoutP 32MB | gates 128KB  (~109 MB)
  char* ws = (char*)d_ws;
  __bf16* xbp   = (__bf16*)ws;
  __bf16* WinP  = (__bf16*)(ws + (size_t)8388608);
  __bf16* WoutP = (__bf16*)(ws + (size_t)8388608 + 67108864);
  float*  gates = (float*) (ws + (size_t)8388608 + 67108864 + 33554432);

  const int n4 = T_TOK * Hd / 4;
  x_to_bf16_kernel<<<n4 / 256, 256, 0, stream>>>(x, xbp, n4);
  pack_win_kernel <<<(Ed * Hd * 2 * Id) / 256, 256, 0, stream>>>(W_in, WinP);
  pack_wout_kernel<<<(Ed * Id * Hd) / 256, 256, 0, stream>>>(W_out, WoutP);
  moe_router_kernel<<<T_TOK / 32, 256, 0, stream>>>(x, Wg, gates);
  moe_ffn_fused_kernel<<<T_TOK / MT, 256, 0, stream>>>(xbp, WinP, WoutP, gates, out);
}
